// Attention_23905787969692
// MI455X (gfx1250) — matrix operand order repaired and checked by reading, not compiled, not run on hardware
//
#include <hip/hip_runtime.h>
#include <hip/hip_bf16.h>

// Problem constants (match reference)
#define BB 4
#define SS 2048
#define EE 1024
#define DD 1024

typedef __bf16 bf16;
typedef __attribute__((ext_vector_type(16))) __bf16 bf16x16;
typedef __attribute__((ext_vector_type(8)))  __bf16 bf16x8;
typedef __attribute__((ext_vector_type(8)))  float   f32x8;

#define WMMA_BF16(a, b, c) \
  __builtin_amdgcn_wmma_f32_16x16x32_bf16(false, (a), false, (b), (short)0, (c), false, false)

#define ASYNC_LDS_B128(ldsoff, gptr) \
  asm volatile("global_load_async_to_lds_b128 %0, %1, off" :: "v"(ldsoff), "v"(gptr) : "memory")
#define WAIT_ASYNC(n)  asm volatile("s_wait_asynccnt " #n ::: "memory")
#define WAIT_DS0()     asm volatile("s_wait_dscnt 0x0" ::: "memory")

static __device__ inline unsigned lds_off(const void* p) {
  return (unsigned)(uintptr_t)p;                 // low 32 bits = LDS byte offset
}

// A-matrix fragment (16x32 bf16): lane holds row m = lane&15,
// frag[0..7]  = K kb..kb+7, frag[8..15] = K kb+16..kb+23, kb = 8*(lane>>4).
static __device__ inline bf16x16 load_a_frag(const bf16* base, int stride, int lane) {
  const int m  = lane & 15;
  const int kb = (lane >> 4) << 3;               // 0 or 8
  const bf16* p = base + (size_t)m * stride + kb;
  bf16x8 lo = *(const bf16x8*)(p);
  bf16x8 hi = *(const bf16x8*)(p + 16);
  return __builtin_shufflevector(lo, hi, 0,1,2,3,4,5,6,7,8,9,10,11,12,13,14,15);
}

// B-matrix fragment (32x16 bf16): lane holds column n = lane&15,
// frag[0..7]  = K kb..kb+7, frag[8..15] = K kb+16..kb+23, kb = 8*(lane>>4).
static __device__ inline bf16x16 load_b_frag(const bf16* base, int stride, int lane) {
  const int n  = lane & 15;
  const int kb = (lane >> 4) << 3;               // 0 or 8
  const bf16* p = base + (size_t)n * stride + kb;
  bf16x8 lo = *(const bf16x8*)(p);
  bf16x8 hi = *(const bf16x8*)(p + 16);
  return __builtin_shufflevector(lo, hi, 0,1,2,3,4,5,6,7,8,9,10,11,12,13,14,15);
}

// ---------------------------------------------------------------- converts
__global__ __launch_bounds__(256)
void cvt_x_kernel(const float* __restrict__ X, bf16* __restrict__ Xbf, int n) {
  int i4 = (blockIdx.x * 256 + threadIdx.x) * 4;
  if (i4 + 3 < n) {
    float4 v = *(const float4*)(X + i4);
    Xbf[i4 + 0] = (bf16)v.x;
    Xbf[i4 + 1] = (bf16)v.y;
    Xbf[i4 + 2] = (bf16)v.z;
    Xbf[i4 + 3] = (bf16)v.w;
  }
}

// Pack W[d][e] (row-major D x E) into Wpk[mat][kt][d][j] with e = kt*32 + j,
// so staged tiles / B-fragments are K-contiguous.
__global__ __launch_bounds__(256)
void pack_w_kernel(const float* __restrict__ Wq, const float* __restrict__ Wk,
                   const float* __restrict__ Wv, bf16* __restrict__ Wpk) {
  const int mat = blockIdx.y;
  const float* W = (mat == 0) ? Wq : (mat == 1) ? Wk : Wv;
  int i = blockIdx.x * 256 + threadIdx.x;        // over D*E
  if (i < DD * EE) {
    int d = i / EE, e = i % EE;
    int kt = e >> 5, j = e & 31;
    Wpk[(size_t)mat * DD * EE + ((size_t)kt * DD + d) * 32 + j] = (bf16)W[i];
  }
}

// ---------------------------------------------------------------- QKV GEMM
// LDS-tiled 128x128 block GEMM with async double-buffered stages.
// grid: (M/128, N/128, 3). Block = 256 threads = 8 waves; wave w computes
// rows [w*16, w*16+16) x all 128 cols. Per K-step (32 deep): stage 128x32 X
// and 128x32 W tiles into LDS via ASYNCcnt ping-pong (4 b128 loads/thread).
__global__ __launch_bounds__(256)
void qkv_kernel(const bf16* __restrict__ Xbf, const bf16* __restrict__ Wpk,
                const float* __restrict__ bq, const float* __restrict__ bk,
                const float* __restrict__ bv,
                bf16* __restrict__ Qbf, bf16* __restrict__ Kbf, bf16* __restrict__ Vt) {
  __shared__ __align__(16) bf16 As[2][128][40];  // +pad: aligned, conflict-free
  __shared__ __align__(16) bf16 Ws[2][128][40];

  const int lane = threadIdx.x & 31;
  const int wave = threadIdx.x >> 5;
  const int row0 = blockIdx.x * 128;             // rows of [B*S, E]
  const int n0   = blockIdx.y * 128;
  const int mat  = blockIdx.z;                   // 0:Q 1:K 2:V
  const bf16* W  = Wpk + (size_t)mat * DD * EE;
  const float* bias = (mat == 0) ? bq : (mat == 1) ? bk : bv;

  auto stage = [&](int buf, int kt) {            // 4 async b128 loads / thread
#pragma unroll
    for (int t = 0; t < 2; ++t) {
      const int chunk = threadIdx.x + t * 256;   // 0..511
      const int r = chunk >> 2, part = chunk & 3;
      const bf16* gA = Xbf + (size_t)(row0 + r) * EE + kt * 32 + part * 8;
      ASYNC_LDS_B128(lds_off(&As[buf][r][part * 8]), gA);
      const bf16* gW = W + ((size_t)kt * DD + n0 + r) * 32 + part * 8;
      ASYNC_LDS_B128(lds_off(&Ws[buf][r][part * 8]), gW);
    }
  };

  f32x8 acc[8];
#pragma unroll
  for (int i = 0; i < 8; ++i) acc[i] = (f32x8){0,0,0,0,0,0,0,0};

  const int KT = EE / 32;                        // 32
  stage(0, 0);
  WAIT_ASYNC(0x0);
  __syncthreads();
  stage(1, 1);

  for (int kt = 0; kt < KT; ++kt) {
    const int cur = kt & 1;
    bf16x16 a = load_a_frag(&As[cur][wave * 16][0], 40, lane);
#pragma unroll
    for (int sub = 0; sub < 8; ++sub) {
      bf16x16 bfr = load_b_frag(&Ws[cur][sub * 16][0], 40, lane);
      acc[sub] = WMMA_BF16(a, bfr, acc[sub]);
    }
    if (kt < KT - 1) {
      __syncthreads();                           // everyone done reading buf cur
      if (kt + 2 < KT) {
        stage(cur, kt + 2);                      // refill just-freed buffer
        WAIT_ASYNC(0x4);                         // prior stage done, new in flight
      } else {
        WAIT_ASYNC(0x0);
      }
      __syncthreads();                           // next buffer valid for all
    }
  }

  const int hi = lane >> 4;
  const int nl = lane & 15;
#pragma unroll
  for (int sub = 0; sub < 8; ++sub) {
    const int n = n0 + sub * 16 + nl;
    const float bb = bias[n];
#pragma unroll
    for (int v = 0; v < 8; ++v) {
      const int m = v + 8 * hi;
      const int r = row0 + wave * 16 + m;        // global row in [B*S)
      const float val = acc[sub][v] + bb;
      if (mat == 2) {                            // V stored transposed: Vt[b][n][s]
        const int b_ = r / SS, s_ = r % SS;
        Vt[((size_t)(b_ * DD + n)) * SS + s_] = (bf16)val;
      } else {
        bf16* dst = (mat == 0) ? Qbf : Kbf;
        dst[(size_t)r * DD + n] = (bf16)val;
      }
    }
  }
}

// ---------------------------------------------------------------- fused attention
// grid: (S/16, B). Block = 256 threads = 8 waves per 16-query tile.
// Q tile (32 KB, contiguous) staged once via async loads. K stream staged per
// wave into private LDS ping-pong buffers (no barriers, ASYNCcnt-only sync).
__global__ __launch_bounds__(256)
void attn_kernel(const bf16* __restrict__ Qbf, const bf16* __restrict__ Kbf,
                 const bf16* __restrict__ Vt, float* __restrict__ Out) {
  __shared__ __align__(16) bf16 Qs[16][EE];      // 32 KB staged Q tile
  __shared__ __align__(16) bf16 Ks[8][2][16][40];// wave-private K ping-pong
  __shared__ float sm_m[8][16];
  __shared__ float sm_l[8][16];
  __shared__ __align__(16) bf16 sp[16][128];     // 8 P-tiles of 16x16 per group

  const int lane = threadIdx.x & 31;
  const int wave = threadIdx.x >> 5;
  const int qt   = blockIdx.x;
  const int b    = blockIdx.y;
  const int q0   = qt * 16;
  const int nl   = lane & 15;
  const int hi   = lane >> 4;
  const float scale = 0.03125f;                  // 1/sqrt(1024)

  const bf16* Qb = Qbf + (size_t)(b * SS + q0) * DD;
  const bf16* Kb = Kbf + (size_t)b * SS * DD;
  const bf16* Vb = Vt  + (size_t)b * DD * SS;

  // ---- async copy of the contiguous 32 KB Q tile into LDS
  {
    const char* gsrc = (const char*)Qb;
#pragma unroll
    for (int it = 0; it < 8; ++it) {
      unsigned off = (threadIdx.x + it * 256) * 16;
      ASYNC_LDS_B128(lds_off(&Qs[0][0]) + off, gsrc + off);
    }
    WAIT_ASYNC(0x0);
  }
  __syncthreads();

  // Wave-private staging of one 16x32 K slice (2 async b128 loads / lane).
  auto kstage = [&](int buf, const bf16* kp, int dk) {
#pragma unroll
    for (int t = 0; t < 2; ++t) {
      const int chunk = lane + t * 32;           // 0..63
      const int r = chunk >> 2, part = chunk & 3;
      const bf16* g = kp + (size_t)r * DD + dk * 32 + part * 8;
      ASYNC_LDS_B128(lds_off(&Ks[wave][buf][r][part * 8]), g);
    }
  };

  const int DK = DD / 32;                        // 32
  // Q @ K^T for key tile kt -> 16x16 f32 scores (wave-private pipeline).
  auto score_tile = [&](int kt) -> f32x8 {
    const bf16* kp = Kb + (size_t)(kt * 16) * DD;
    f32x8 acc = (f32x8){0,0,0,0,0,0,0,0};
    kstage(0, kp, 0);
    WAIT_ASYNC(0x0);
    kstage(1, kp, 1);
    for (int dk = 0; dk < DK; ++dk) {
      const int cur = dk & 1;
      bf16x16 a  = load_a_frag(&Qs[0][dk * 32], EE, lane);
      bf16x16 kf = load_b_frag(&Ks[wave][cur][0][0], 40, lane);
      acc = WMMA_BF16(a, kf, acc);
      if (dk < DK - 1) {
        WAIT_DS0();                              // ds reads landed before rewrite
        if (dk + 2 < DK) {
          kstage(cur, kp, dk + 2);
          WAIT_ASYNC(0x2);                       // prior slice done, new in flight
        } else {
          WAIT_ASYNC(0x0);
        }
      }
    }
    return acc;
  };

  const int nkt = qt + 1;                        // causal key tiles

  float m_run[8], l_run[8];
#pragma unroll
  for (int v = 0; v < 8; ++v) { m_run[v] = -3.0e38f; l_run[v] = 0.0f; }

  // ---------- sweep 1: row statistics ----------
  for (int kt = wave; kt < nkt; kt += 8) {
    f32x8 acc = score_tile(kt);
    const bool diag = (kt == qt);
#pragma unroll
    for (int v = 0; v < 8; ++v) {
      const int m = v + 8 * hi;
      float s = acc[v] * scale;
      if (diag && nl > m) s = -3.0e38f;          // causal mask inside diagonal tile
      float t = s;                               // row max across the 16-lane half
      t = fmaxf(t, __shfl_xor(t, 1, 32));
      t = fmaxf(t, __shfl_xor(t, 2, 32));
      t = fmaxf(t, __shfl_xor(t, 4, 32));
      t = fmaxf(t, __shfl_xor(t, 8, 32));
      const float mnew = fmaxf(m_run[v], t);
      float e = __expf(s - mnew);
      e += __shfl_xor(e, 1, 32);
      e += __shfl_xor(e, 2, 32);
      e += __shfl_xor(e, 4, 32);
      e += __shfl_xor(e, 8, 32);
      l_run[v] = l_run[v] * __expf(m_run[v] - mnew) + e;
      m_run[v] = mnew;
    }
  }

  if (nl == 0) {
#pragma unroll
    for (int v = 0; v < 8; ++v) {
      sm_m[wave][v + 8 * hi] = m_run[v];
      sm_l[wave][v + 8 * hi] = l_run[v];
    }
  }
  __syncthreads();

  float Mrow[8], Linv[8];
#pragma unroll
  for (int v = 0; v < 8; ++v) {
    const int m = v + 8 * hi;
    float Mx = sm_m[0][m];
#pragma unroll
    for (int w = 1; w < 8; ++w) Mx = fmaxf(Mx, sm_m[w][m]);
    float Ls = 0.0f;
#pragma unroll
    for (int w = 0; w < 8; ++w) Ls += sm_l[w][m] * __expf(sm_m[w][m] - Mx);
    Mrow[v] = Mx;
    Linv[v] = 1.0f / Ls;                         // diagonal guarantees Ls >= 1
  }

  // ---------- sweep 2: P tiles -> LDS -> O += P @ Vt ----------
  f32x8 oacc[8];
#pragma unroll
  for (int i = 0; i < 8; ++i) oacc[i] = (f32x8){0,0,0,0,0,0,0,0};
  const int nbase = wave * 128;
  const int ngrp  = (nkt + 7) >> 3;

  for (int g = 0; g < ngrp; ++g) {
    const int kt = g * 8 + wave;
    f32x8 acc = (f32x8){0,0,0,0,0,0,0,0};
    if (kt < nkt) acc = score_tile(kt);
    __syncthreads();                             // previous group's sp reads done
    const bool diag = (kt == qt);
#pragma unroll
    for (int v = 0; v < 8; ++v) {
      const int m = v + 8 * hi;
      float p = 0.0f;
      if (kt < nkt) {
        float s = acc[v] * scale;
        p = (diag && nl > m) ? 0.0f : __expf(s - Mrow[v]);
      }
      sp[m][wave * 16 + nl] = (bf16)p;           // C-layout scatter into LDS
    }
    __syncthreads();                             // sp ready for all waves

    const int key0g = g * 128;
#pragma unroll
    for (int pr = 0; pr < 4; ++pr) {             // pairs of P tiles => K=32 keys
      bf16x16 a = load_a_frag(&sp[0][pr * 32], 128, lane);
#pragma unroll
      for (int sub = 0; sub < 8; ++sub) {
        bf16x16 vb = load_b_frag(
            Vb + (size_t)(nbase + sub * 16) * SS + key0g + pr * 32, SS, lane);
        oacc[sub] = WMMA_BF16(a, vb, oacc[sub]);
      }
    }
  }

  float* Ob = Out + (size_t)(b * SS + q0) * DD;
#pragma unroll
  for (int sub = 0; sub < 8; ++sub) {
    const int n = nbase + sub * 16 + nl;
#pragma unroll
    for (int v = 0; v < 8; ++v) {
      const int m = v + 8 * hi;
      Ob[(size_t)m * DD + n] = oacc[sub][v] * Linv[v];
    }
  }
}

// ---------------------------------------------------------------- launcher
extern "C" void kernel_launch(void* const* d_in, const int* in_sizes, int n_in,
                              void* d_out, int out_size, void* d_ws, size_t ws_size,
                              hipStream_t stream) {
  const float* X  = (const float*)d_in[0];
  const float* Wq = (const float*)d_in[1];
  const float* bq = (const float*)d_in[2];
  const float* Wk = (const float*)d_in[3];
  const float* bk = (const float*)d_in[4];
  const float* Wv = (const float*)d_in[5];
  const float* bv = (const float*)d_in[6];
  // d_in[7] (mask) unused: causal mask is computed analytically.
  float* Out = (float*)d_out;

  char* ws = (char*)d_ws;
  bf16* Xbf = (bf16*)ws; ws += (size_t)BB * SS * EE * 2;   // 16 MB
  bf16* Wpk = (bf16*)ws; ws += (size_t)3 * EE * DD * 2;    // 6 MB
  bf16* Qbf = (bf16*)ws; ws += (size_t)BB * SS * DD * 2;   // 16 MB
  bf16* Kbf = (bf16*)ws; ws += (size_t)BB * SS * DD * 2;   // 16 MB
  bf16* Vt  = (bf16*)ws; ws += (size_t)BB * DD * SS * 2;   // 16 MB (transposed V)

  const int nX = BB * SS * EE;
  cvt_x_kernel<<<dim3((nX / 4 + 255) / 256), 256, 0, stream>>>(X, Xbf, nX);
  pack_w_kernel<<<dim3((DD * EE + 255) / 256, 3), 256, 0, stream>>>(Wq, Wk, Wv, Wpk);
  qkv_kernel<<<dim3(BB * SS / 16 / 8, DD / 128, 3), 256, 0, stream>>>(Xbf, Wpk, bq, bk, bv,
                                                                      Qbf, Kbf, Vt);
  attn_kernel<<<dim3(SS / 16, BB), 256, 0, stream>>>(Qbf, Kbf, Vt, Out);
}
